// DANet_64029372448974
// MI455X (gfx1250) — matrix operand order repaired and checked by reading, not compiled, not run on hardware
//
#include <hip/hip_runtime.h>

// ---------------------------------------------------------------------------
// DANet head for MI455X (gfx1250): all matmul-like work through
// v_wmma_f32_16x16x32_bf16 (bf16 A/B, f32 accumulate), f32 elsewhere.
// Block tile 128x128, K-step 32; LDS B-tile stored transposed so all WMMA
// fragment loads are ds_load_b128.
// ---------------------------------------------------------------------------

typedef __attribute__((ext_vector_type(16))) __bf16 v16bf;
typedef __attribute__((ext_vector_type(8)))  __bf16 v8bf;
typedef __attribute__((ext_vector_type(8)))  float  v8f;
typedef long long i64;

// Problem constants
#define BATCH 2
#define CIN   2048
#define CMID  512
#define COUT  19
#define HH    48
#define HW    2304          // 48*48
#define NTOT  4608          // BATCH*HW

// GEMM tiling: block = 128(M) x 128(N), K-step 32, 256 threads = 8 waves in
// 4(M) x 2(N); each wave owns a 32x64 subtile = 2x4 WMMA tiles.
#define TBM 128
#define TBN 128
#define TBK 32

// ---- WMMA fragment loads from LDS (ISA 7.12.2 layouts, wave32) -------------
// A 16x32 bf16 row-major tile (row stride TBK): lane 0-15 row M=lane holds
// K={ko..ko+7, 16+ko..16+ko+7} with ko=0; lanes 16-31 ko=8.  Two b128 loads.
__device__ inline v16bf frag_a(const __bf16* base, int lane) {
  int m  = lane & 15;
  int ko = (lane & 16) ? 8 : 0;
  const __bf16* r = base + m * TBK + ko;
  v8bf lo = *(const v8bf*)(r);        // K = ko .. ko+7
  v8bf hi = *(const v8bf*)(r + 16);   // K = 16+ko .. 16+ko+7
  return __builtin_shufflevector(lo, hi, 0,1,2,3,4,5,6,7,8,9,10,11,12,13,14,15);
}
// B 32x16 bf16 from transposed tile sBt[n][k] (row stride TBK): lanes 0-15
// col N=lane hold K={0..7,16..23}, lanes 16-31 hold K={8..15,24..31}.  Two b128 loads.
__device__ inline v16bf frag_b(const __bf16* base, int lane) {
  int n  = lane & 15;
  int kb = (lane & 16) ? 8 : 0;
  const __bf16* r = base + n * TBK + kb;
  v8bf lo = *(const v8bf*)(r);
  v8bf hi = *(const v8bf*)(r + 16);
  return __builtin_shufflevector(lo, hi, 0,1,2,3,4,5,6,7,8,9,10,11,12,13,14,15);
}

// ---------------------------------------------------------------------------
// Generic batched strided GEMM: C[z][m][n] = sum_k A[z](m,k)*B[z](k,n)+bias[m]
// A element (m,k) at A[z*abz + m*ars + k*acs] (bf16); same scheme for B.
// Vector staging fast paths for acs==1 / bcs==1 / brs==1 (uniform branches).
// ---------------------------------------------------------------------------
__global__ __launch_bounds__(256)
void k_gemm_bf16(int M, int N, int K,
                 const __bf16* __restrict__ A, i64 ars, i64 acs, i64 abz,
                 const __bf16* __restrict__ Bm, i64 brs, i64 bcs, i64 bbz,
                 float* __restrict__ Cf, i64 ldc, i64 cbz,
                 __bf16* __restrict__ Cb, i64 ldcb, i64 cbbz,
                 const float* __restrict__ bias)
{
  __shared__ __align__(16) __bf16 sA [TBM * TBK];   // [m][k]
  __shared__ __align__(16) __bf16 sBt[TBN * TBK];   // [n][k] (transposed)
  const int tid  = threadIdx.x;
  const int lane = tid & 31;
  const int wid  = tid >> 5;
  const int m0   = blockIdx.y * TBM;
  const int n0   = blockIdx.x * TBN;
  const i64 z    = blockIdx.z;
  A  += z * abz;
  Bm += z * bbz;

  const bool a_vec = (acs == 1);
  const bool b_row = (brs == 1);
  const bool b_col = (bcs == 1);

  const int wm = (wid >> 1) * 32;   // 0,32,64,96
  const int wn = (wid & 1) * 64;    // 0,64
  v8f acc[2][4] = {};

  for (int k0 = 0; k0 < K; k0 += TBK) {
    // ---- stage A tile (128x32 row-major) ----
    if (a_vec) {
      int m  = tid >> 1;
      int k  = (tid & 1) * 16;
      int gm = m0 + m;
      v8bf u0 = {}, u1 = {};
      if (gm < M) {
        const __bf16* g = A + (i64)gm * ars + (k0 + k);
        u0 = *(const v8bf*)(g);
        u1 = *(const v8bf*)(g + 8);
      }
      *(v8bf*)(&sA[m * TBK + k])     = u0;
      *(v8bf*)(&sA[m * TBK + k + 8]) = u1;
    } else {
#pragma unroll
      for (int e = 0; e < (TBM * TBK) / 256; ++e) {
        int i = e * 256 + tid;
        int m = i >> 5, k = i & 31;
        int gm = m0 + m, gk = k0 + k;
        __bf16 v = (__bf16)0.0f;
        if (gm < M && gk < K) v = A[(i64)gm * ars + (i64)gk * acs];
        sA[m * TBK + k] = v;
      }
    }
    // ---- stage B tile transposed into sBt[n][k] ----
    if (b_row) {
      // B(k,n) contiguous in k: fully vector (global b128 -> LDS b128)
      int n  = tid >> 1;
      int k  = (tid & 1) * 16;
      int gn = n0 + n;
      v8bf u0 = {}, u1 = {};
      if (gn < N) {
        const __bf16* g = Bm + (i64)(k0 + k) + (i64)gn * bcs;
        u0 = *(const v8bf*)(g);
        u1 = *(const v8bf*)(g + 8);
      }
      *(v8bf*)(&sBt[n * TBK + k])     = u0;
      *(v8bf*)(&sBt[n * TBK + k + 8]) = u1;
    } else if (b_col) {
      // B(k,n) contiguous in n: vector global load, scalar transpose store
      int k  = tid >> 3;            // 0..31
      int nc = (tid & 7) * 16;      // 0..112
      int gk = k0 + k;
      v8bf u0 = {}, u1 = {};
      if (gk < K && (n0 + nc) < N) {
        const __bf16* g = Bm + (i64)gk * brs + (n0 + nc);
        u0 = *(const v8bf*)(g);
        u1 = *(const v8bf*)(g + 8);
      }
#pragma unroll
      for (int j = 0; j < 8; ++j) sBt[(nc + j) * TBK + k] = u0[j];
#pragma unroll
      for (int j = 0; j < 8; ++j) sBt[(nc + 8 + j) * TBK + k] = u1[j];
    } else {
#pragma unroll
      for (int e = 0; e < (TBN * TBK) / 256; ++e) {
        int i = e * 256 + tid;
        int n = i >> 5, k = i & 31;
        int gk = k0 + k, gn = n0 + n;
        __bf16 v = (__bf16)0.0f;
        if (gk < K && gn < N) v = Bm[(i64)gk * brs + (i64)gn * bcs];
        sBt[n * TBK + k] = v;
      }
    }
    if (k0 + TBK < K) {
      __builtin_prefetch(&A[(i64)(m0 + (tid >> 1)) * ars + (i64)(k0 + TBK) * acs], 0, 1);
    }
    __syncthreads();

    v16bf a0 = frag_a(&sA[(wm)      * TBK], lane);
    v16bf a1 = frag_a(&sA[(wm + 16) * TBK], lane);
    v16bf b0 = frag_b(&sBt[(wn)      * TBK], lane);
    v16bf b1 = frag_b(&sBt[(wn + 16) * TBK], lane);
    v16bf b2 = frag_b(&sBt[(wn + 32) * TBK], lane);
    v16bf b3 = frag_b(&sBt[(wn + 48) * TBK], lane);
    acc[0][0] = __builtin_amdgcn_wmma_f32_16x16x32_bf16(false, a0, false, b0, (short)0, acc[0][0], false, false);
    acc[0][1] = __builtin_amdgcn_wmma_f32_16x16x32_bf16(false, a0, false, b1, (short)0, acc[0][1], false, false);
    acc[0][2] = __builtin_amdgcn_wmma_f32_16x16x32_bf16(false, a0, false, b2, (short)0, acc[0][2], false, false);
    acc[0][3] = __builtin_amdgcn_wmma_f32_16x16x32_bf16(false, a0, false, b3, (short)0, acc[0][3], false, false);
    acc[1][0] = __builtin_amdgcn_wmma_f32_16x16x32_bf16(false, a1, false, b0, (short)0, acc[1][0], false, false);
    acc[1][1] = __builtin_amdgcn_wmma_f32_16x16x32_bf16(false, a1, false, b1, (short)0, acc[1][1], false, false);
    acc[1][2] = __builtin_amdgcn_wmma_f32_16x16x32_bf16(false, a1, false, b2, (short)0, acc[1][2], false, false);
    acc[1][3] = __builtin_amdgcn_wmma_f32_16x16x32_bf16(false, a1, false, b3, (short)0, acc[1][3], false, false);
    __syncthreads();
  }

  // C/D layout: lanes 0-15 -> (M=r, N=lane); lanes 16-31 -> (M=8+r, N=lane-16)
  const int ncol = lane & 15;
  const int mrow = (lane >> 4) * 8;
#pragma unroll
  for (int ti = 0; ti < 2; ++ti)
#pragma unroll
    for (int tj = 0; tj < 4; ++tj)
#pragma unroll
      for (int r = 0; r < 8; ++r) {
        int gm = m0 + wm + ti * 16 + mrow + r;
        int gn = n0 + wn + tj * 16 + ncol;
        if (gm < M && gn < N) {
          float v = acc[ti][tj][r];
          if (bias) v += bias[gm];
          if (Cf) Cf[z * cbz + (i64)gm * ldc + gn] = v;
          if (Cb) Cb[z * cbbz + (i64)gm * ldcb + gn] = (__bf16)v;
        }
      }
}

// ---------------------------------------------------------------------------
// Implicit-GEMM 3x3 SAME conv: Y[cout][b*HW+hw] = sum_{cin,kh,kw} W*X
// W: bf16 [Cout][Cin*9];  X element (b, cin, ih, iw) at
//   X[cin*xcs + b*xbz + ih*48 + iw]   (covers NCHW and [C][B][HW] layouts)
// ---------------------------------------------------------------------------
__global__ __launch_bounds__(256)
void k_conv3x3_wmma(int Cout, int Cinp,
                    const __bf16* __restrict__ W,
                    const __bf16* __restrict__ X, i64 xcs, i64 xbz,
                    float* __restrict__ Y)
{
  __shared__ __align__(16) __bf16 sA [TBM * TBK];
  __shared__ __align__(16) __bf16 sBt[TBN * TBK];
  const int tid  = threadIdx.x;
  const int lane = tid & 31;
  const int wid  = tid >> 5;
  const int m0   = blockIdx.y * TBM;
  const int n0   = blockIdx.x * TBN;
  const int K    = Cinp * 9;

  const int wm = (wid >> 1) * 32;
  const int wn = (wid & 1) * 64;
  v8f acc[2][4] = {};

  __builtin_amdgcn_s_wait_tensorcnt(0);

  for (int k0 = 0; k0 < K; k0 += TBK) {
    // weights tile (row-major, contiguous K) -> vector staging
    {
      int m = tid >> 1;
      int k = (tid & 1) * 16;
      const __bf16* g = W + (i64)(m0 + m) * K + (k0 + k);
      v8bf u0 = *(const v8bf*)(g);
      v8bf u1 = *(const v8bf*)(g + 8);
      *(v8bf*)(&sA[m * TBK + k])     = u0;
      *(v8bf*)(&sA[m * TBK + k + 8]) = u1;
    }
    // im2col tile -> sBt[n][k] (scalar gather; halo handling)
#pragma unroll
    for (int e = 0; e < (TBN * TBK) / 256; ++e) {
      int i  = e * 256 + tid;
      int kl = i >> 7;            // 0..31
      int nl = i & 127;
      int k  = k0 + kl;
      int gn = n0 + nl;
      int b  = gn / HW;
      int hw = gn - b * HW;
      int h  = hw / HH;
      int w  = hw - h * HH;
      int cin = k / 9;
      int r   = k - cin * 9;
      int kh  = r / 3;
      int kw  = r - kh * 3;
      int ih = h + kh - 1, iw = w + kw - 1;
      __bf16 v = (__bf16)0.0f;
      if (ih >= 0 && ih < HH && iw >= 0 && iw < HH)
        v = X[(i64)cin * xcs + (i64)b * xbz + ih * HH + iw];
      sBt[nl * TBK + kl] = v;
    }
    __syncthreads();

    v16bf a0 = frag_a(&sA[(wm)      * TBK], lane);
    v16bf a1 = frag_a(&sA[(wm + 16) * TBK], lane);
    v16bf b0 = frag_b(&sBt[(wn)      * TBK], lane);
    v16bf b1 = frag_b(&sBt[(wn + 16) * TBK], lane);
    v16bf b2 = frag_b(&sBt[(wn + 32) * TBK], lane);
    v16bf b3 = frag_b(&sBt[(wn + 48) * TBK], lane);
    acc[0][0] = __builtin_amdgcn_wmma_f32_16x16x32_bf16(false, a0, false, b0, (short)0, acc[0][0], false, false);
    acc[0][1] = __builtin_amdgcn_wmma_f32_16x16x32_bf16(false, a0, false, b1, (short)0, acc[0][1], false, false);
    acc[0][2] = __builtin_amdgcn_wmma_f32_16x16x32_bf16(false, a0, false, b2, (short)0, acc[0][2], false, false);
    acc[0][3] = __builtin_amdgcn_wmma_f32_16x16x32_bf16(false, a0, false, b3, (short)0, acc[0][3], false, false);
    acc[1][0] = __builtin_amdgcn_wmma_f32_16x16x32_bf16(false, a1, false, b0, (short)0, acc[1][0], false, false);
    acc[1][1] = __builtin_amdgcn_wmma_f32_16x16x32_bf16(false, a1, false, b1, (short)0, acc[1][1], false, false);
    acc[1][2] = __builtin_amdgcn_wmma_f32_16x16x32_bf16(false, a1, false, b2, (short)0, acc[1][2], false, false);
    acc[1][3] = __builtin_amdgcn_wmma_f32_16x16x32_bf16(false, a1, false, b3, (short)0, acc[1][3], false, false);
    __syncthreads();
  }

  const int ncol = lane & 15;
  const int mrow = (lane >> 4) * 8;
#pragma unroll
  for (int ti = 0; ti < 2; ++ti)
#pragma unroll
    for (int tj = 0; tj < 4; ++tj)
#pragma unroll
      for (int r = 0; r < 8; ++r) {
        int gm = m0 + wm + ti * 16 + mrow + r;
        int gn = n0 + wn + tj * 16 + ncol;
        if (gm < Cout && gn < NTOT)
          Y[(i64)gm * NTOT + gn] = acc[ti][tj][r];
      }
}

// ---------------------------------------------------------------------------
// Elementwise / reduction helpers (f32 path)
// ---------------------------------------------------------------------------
__global__ void k_cvt_bf16(const float* __restrict__ s, __bf16* __restrict__ d, i64 n) {
  i64 i = (i64)blockIdx.x * blockDim.x + threadIdx.x;
  if (i < n) d[i] = (__bf16)s[i];
}

// Training-mode BN over (N,H,W) per channel + ReLU; in-place f32 + bf16 copy.
__global__ __launch_bounds__(256)
void k_bn_relu(float* __restrict__ Y, const float* __restrict__ G,
               const float* __restrict__ Bb, __bf16* __restrict__ Yb, int Ntot)
{
  int c = blockIdx.x;
  float* p = Y + (i64)c * Ntot;
  __bf16* ob = Yb + (i64)c * Ntot;
  __shared__ float s1[256], s2[256];
  int t = threadIdx.x;
  float sum = 0.f, sq = 0.f;
  for (int i = t; i < Ntot; i += 256) { float v = p[i]; sum += v; sq += v * v; }
  s1[t] = sum; s2[t] = sq; __syncthreads();
  for (int off = 128; off > 0; off >>= 1) {
    if (t < off) { s1[t] += s1[t + off]; s2[t] += s2[t + off]; }
    __syncthreads();
  }
  float mu  = s1[0] / Ntot;
  float var = s2[0] / Ntot - mu * mu;
  float inv = rsqrtf(var + 1e-5f);
  float g = G[c], b = Bb[c];
  for (int i = t; i < Ntot; i += 256) {
    float v = (p[i] - mu) * inv * g + b;
    v = fmaxf(v, 0.0f);
    p[i] = v;
    ob[i] = (__bf16)v;
  }
}

// Row softmax of sign*x (sign=-1 implements CAM's max-x trick), bf16 out.
__global__ __launch_bounds__(256)
void k_softmax_rows(const float* __restrict__ E, __bf16* __restrict__ A, int L, float sign)
{
  i64 row = blockIdx.x;
  const float* p = E + row * (i64)L;
  __bf16* o = A + row * (i64)L;
  __shared__ float red[256];
  int t = threadIdx.x;
  float m = -3.4e38f;
  for (int i = t; i < L; i += 256) m = fmaxf(m, sign * p[i]);
  red[t] = m; __syncthreads();
  for (int off = 128; off > 0; off >>= 1) {
    if (t < off) red[t] = fmaxf(red[t], red[t + off]);
    __syncthreads();
  }
  m = red[0]; __syncthreads();
  float s = 0.f;
  for (int i = t; i < L; i += 256) s += __expf(sign * p[i] - m);
  red[t] = s; __syncthreads();
  for (int off = 128; off > 0; off >>= 1) {
    if (t < off) red[t] += red[t + off];
    __syncthreads();
  }
  float inv = 1.0f / red[0];
  for (int i = t; i < L; i += 256) o[i] = (__bf16)(__expf(sign * p[i] - m) * inv);
}

__global__ void k_residual(const float* __restrict__ ao, const float* __restrict__ ft,
                           const float* __restrict__ gamma,
                           float* __restrict__ of, __bf16* __restrict__ ob, i64 n)
{
  i64 i = (i64)blockIdx.x * blockDim.x + threadIdx.x;
  if (i < n) {
    float v = gamma[0] * ao[i] + ft[i];
    of[i] = v;
    ob[i] = (__bf16)v;
  }
}

__global__ void k_add_bf(const float* __restrict__ a, const float* __restrict__ b,
                         __bf16* __restrict__ o, i64 n)
{
  i64 i = (i64)blockIdx.x * blockDim.x + threadIdx.x;
  if (i < n) o[i] = (__bf16)(a[i] + b[i]);
}

// ---------------------------------------------------------------------------
// Host orchestration
// ---------------------------------------------------------------------------
static inline char* bump(char*& p, size_t bytes) {
  char* r = p;
  p += (bytes + 255) & ~((size_t)255);
  return r;
}

extern "C" void kernel_launch(void* const* d_in, const int* in_sizes, int n_in,
                              void* d_out, int out_size, void* d_ws, size_t ws_size,
                              hipStream_t stream) {
  (void)in_sizes; (void)n_in; (void)out_size; (void)ws_size;
  const float* x    = (const float*)d_in[0];
  const float* w5a  = (const float*)d_in[1];
  const float* g5a  = (const float*)d_in[2];
  const float* b5a  = (const float*)d_in[3];
  const float* w5c  = (const float*)d_in[4];
  const float* g5c  = (const float*)d_in[5];
  const float* b5c  = (const float*)d_in[6];
  const float* wq   = (const float*)d_in[7];
  const float* bq   = (const float*)d_in[8];
  const float* wk   = (const float*)d_in[9];
  const float* bk   = (const float*)d_in[10];
  const float* wv   = (const float*)d_in[11];
  const float* bv   = (const float*)d_in[12];
  const float* gpam = (const float*)d_in[13];
  const float* gcam = (const float*)d_in[14];
  const float* w51  = (const float*)d_in[15];
  const float* g51  = (const float*)d_in[16];
  const float* b51  = (const float*)d_in[17];
  const float* w52  = (const float*)d_in[18];
  const float* g52  = (const float*)d_in[19];
  const float* b52  = (const float*)d_in[20];
  const float* w6   = (const float*)d_in[21];
  const float* b6   = (const float*)d_in[22];
  const float* w7   = (const float*)d_in[23];
  const float* b7   = (const float*)d_in[24];
  const float* w8   = (const float*)d_in[25];
  const float* b8   = (const float*)d_in[26];
  float* out = (float*)d_out;

  // ---- workspace carve ----
  char* p = (char*)d_ws;
  const i64 nX   = (i64)BATCH * CIN * HW;
  const i64 nW3a = (i64)CMID * CIN * 9;
  const i64 nW3b = (i64)CMID * CMID * 9;
  const i64 nAct = (i64)CMID * NTOT;
  const i64 nQK  = (i64)64 * NTOT;
  const i64 nE   = (i64)BATCH * HW * HW;
  const i64 nEc  = (i64)BATCH * CMID * CMID;

  __bf16* xb     = (__bf16*)bump(p, nX   * 2);
  __bf16* w5a_b  = (__bf16*)bump(p, nW3a * 2);
  __bf16* w5c_b  = (__bf16*)bump(p, nW3a * 2);
  __bf16* w51_b  = (__bf16*)bump(p, nW3b * 2);
  __bf16* w52_b  = (__bf16*)bump(p, nW3b * 2);
  __bf16* wq_b   = (__bf16*)bump(p, (i64)64 * CMID * 2);
  __bf16* wk_b   = (__bf16*)bump(p, (i64)64 * CMID * 2);
  __bf16* wv_b   = (__bf16*)bump(p, (i64)CMID * CMID * 2);
  __bf16* w6_b   = (__bf16*)bump(p, (i64)COUT * CMID * 2);
  __bf16* w7_b   = (__bf16*)bump(p, (i64)COUT * CMID * 2);
  __bf16* w8_b   = (__bf16*)bump(p, (i64)COUT * CMID * 2);
  float*  feat1   = (float*) bump(p, nAct * 4);
  __bf16* feat1_b = (__bf16*)bump(p, nAct * 2);
  float*  feat2   = (float*) bump(p, nAct * 4);
  __bf16* feat2_b = (__bf16*)bump(p, nAct * 2);
  __bf16* q_b     = (__bf16*)bump(p, nQK * 2);
  __bf16* kk_b    = (__bf16*)bump(p, nQK * 2);
  __bf16* v_b     = (__bf16*)bump(p, nAct * 2);
  float*  energy  = (float*) bump(p, nE * 4);
  __bf16* attn    = (__bf16*)bump(p, nE * 2);
  float*  pam_out = (float*) bump(p, nAct * 4);
  float*  sa_feat   = (float*) bump(p, nAct * 4);
  __bf16* sa_feat_b = (__bf16*)bump(p, nAct * 2);
  float*  ec      = (float*) bump(p, nEc * 4);
  __bf16* ac      = (__bf16*)bump(p, nEc * 2);
  float*  cam_out = (float*) bump(p, nAct * 4);
  float*  sc_feat   = (float*) bump(p, nAct * 4);
  __bf16* sc_feat_b = (__bf16*)bump(p, nAct * 2);
  float*  sa_conv   = (float*) bump(p, nAct * 4);
  __bf16* sa_conv_b = (__bf16*)bump(p, nAct * 2);
  float*  sc_conv   = (float*) bump(p, nAct * 4);
  __bf16* sc_conv_b = (__bf16*)bump(p, nAct * 2);
  __bf16* fsum_b    = (__bf16*)bump(p, nAct * 2);

  auto cvt = [&](const float* s, __bf16* d, i64 n) {
    unsigned g = (unsigned)((n + 255) / 256);
    k_cvt_bf16<<<dim3(g), dim3(256), 0, stream>>>(s, d, n);
  };
  auto gemm = [&](int M, int N, int K, int Z,
                  const __bf16* A, i64 ars, i64 acs, i64 abz,
                  const __bf16* Bm, i64 brs, i64 bcs, i64 bbz,
                  float* Cf, i64 ldc, i64 cbz,
                  __bf16* Cb, i64 ldcb, i64 cbbz, const float* bias) {
    dim3 g((N + TBN - 1) / TBN, (M + TBM - 1) / TBM, Z);
    k_gemm_bf16<<<g, dim3(256), 0, stream>>>(M, N, K, A, ars, acs, abz,
                                             Bm, brs, bcs, bbz,
                                             Cf, ldc, cbz, Cb, ldcb, cbbz, bias);
  };
  auto conv3 = [&](int Cout, int Cinp, const __bf16* W, const __bf16* X,
                   i64 xcs, i64 xbz, float* Y) {
    dim3 g(NTOT / TBN, Cout / TBM, 1);
    k_conv3x3_wmma<<<g, dim3(256), 0, stream>>>(Cout, Cinp, W, X, xcs, xbz, Y);
  };

  // ---- precision conversion ----
  cvt(x,   xb,    nX);
  cvt(w5a, w5a_b, nW3a);
  cvt(w5c, w5c_b, nW3a);
  cvt(w51, w51_b, nW3b);
  cvt(w52, w52_b, nW3b);
  cvt(wq,  wq_b,  (i64)64 * CMID);
  cvt(wk,  wk_b,  (i64)64 * CMID);
  cvt(wv,  wv_b,  (i64)CMID * CMID);
  cvt(w6,  w6_b,  (i64)COUT * CMID);
  cvt(w7,  w7_b,  (i64)COUT * CMID);
  cvt(w8,  w8_b,  (i64)COUT * CMID);

  // ===================== PAM branch =====================
  conv3(CMID, CIN, w5a_b, xb, HW, (i64)CIN * HW, feat1);
  k_bn_relu<<<dim3(CMID), dim3(256), 0, stream>>>(feat1, g5a, b5a, feat1_b, NTOT);

  gemm(64,   NTOT, CMID, 1, wq_b, CMID, 1, 0, feat1_b, NTOT, 1, 0,
       nullptr, 0, 0, q_b,  NTOT, 0, bq);
  gemm(64,   NTOT, CMID, 1, wk_b, CMID, 1, 0, feat1_b, NTOT, 1, 0,
       nullptr, 0, 0, kk_b, NTOT, 0, bk);
  gemm(CMID, NTOT, CMID, 1, wv_b, CMID, 1, 0, feat1_b, NTOT, 1, 0,
       nullptr, 0, 0, v_b,  NTOT, 0, bv);

  // energy[b] = q_b^T @ k_b : [HW x HW] per batch
  gemm(HW, HW, 64, BATCH,
       q_b,  1, NTOT, HW,
       kk_b, NTOT, 1, HW,
       energy, HW, (i64)HW * HW, nullptr, 0, 0, nullptr);
  k_softmax_rows<<<dim3(BATCH * HW), dim3(256), 0, stream>>>(energy, attn, HW, 1.0f);

  // pam_out = v_b @ attn_b^T   (B contiguous in k -> fully vector staging)
  gemm(CMID, HW, HW, BATCH,
       v_b, NTOT, 1, HW,
       attn, 1, HW, (i64)HW * HW,
       pam_out, NTOT, HW, nullptr, 0, 0, nullptr);

  k_residual<<<dim3((unsigned)((nAct + 255) / 256)), dim3(256), 0, stream>>>(
      pam_out, feat1, gpam, sa_feat, sa_feat_b, nAct);

  conv3(CMID, CMID, w51_b, sa_feat_b, NTOT, HW, sa_conv);
  k_bn_relu<<<dim3(CMID), dim3(256), 0, stream>>>(sa_conv, g51, b51, sa_conv_b, NTOT);

  // ===================== CAM branch =====================
  conv3(CMID, CIN, w5c_b, xb, HW, (i64)CIN * HW, feat2);
  k_bn_relu<<<dim3(CMID), dim3(256), 0, stream>>>(feat2, g5c, b5c, feat2_b, NTOT);

  gemm(CMID, CMID, HW, BATCH,
       feat2_b, NTOT, 1, HW,
       feat2_b, 1, NTOT, HW,
       ec, CMID, (i64)CMID * CMID, nullptr, 0, 0, nullptr);
  // softmax(max - e) == softmax(-e) rowwise
  k_softmax_rows<<<dim3(BATCH * CMID), dim3(256), 0, stream>>>(ec, ac, CMID, -1.0f);

  gemm(CMID, HW, CMID, BATCH,
       ac, CMID, 1, (i64)CMID * CMID,
       feat2_b, NTOT, 1, HW,
       cam_out, NTOT, HW, nullptr, 0, 0, nullptr);

  k_residual<<<dim3((unsigned)((nAct + 255) / 256)), dim3(256), 0, stream>>>(
      cam_out, feat2, gcam, sc_feat, sc_feat_b, nAct);

  conv3(CMID, CMID, w52_b, sc_feat_b, NTOT, HW, sc_conv);
  k_bn_relu<<<dim3(CMID), dim3(256), 0, stream>>>(sc_conv, g52, b52, sc_conv_b, NTOT);

  // ===================== heads =====================
  float* sasc = out;
  float* sa   = out + (i64)BATCH * COUT * HW;
  float* sc   = out + (i64)2 * BATCH * COUT * HW;

  gemm(COUT, HW, CMID, BATCH,
       w6_b, CMID, 1, 0,
       sa_conv_b, NTOT, 1, HW,
       sa, HW, (i64)COUT * HW, nullptr, 0, 0, b6);
  gemm(COUT, HW, CMID, BATCH,
       w7_b, CMID, 1, 0,
       sc_conv_b, NTOT, 1, HW,
       sc, HW, (i64)COUT * HW, nullptr, 0, 0, b7);

  k_add_bf<<<dim3((unsigned)((nAct + 255) / 256)), dim3(256), 0, stream>>>(
      sa_conv, sc_conv, fsum_b, nAct);
  gemm(COUT, HW, CMID, BATCH,
       w8_b, CMID, 1, 0,
       fsum_b, NTOT, 1, HW,
       sasc, HW, (i64)COUT * HW, nullptr, 0, 0, b8);
}
